// GATLayer_34514357190741
// MI455X (gfx1250) — compile-verified, matrix operand order checked by reading, not run on hardware
//
#include <hip/hip_runtime.h>
#include <hip/hip_bf16.h>
#include <cmath>
#include <cstdint>

// Problem constants (fixed by the reference: B=8, N=2048, Fin=128, H=4, Fout=64)
#define B_    8
#define N_    2048
#define FIN   128
#define H_    4
#define FOUT  64
#define HF    256        // H_*FOUT
#define BH    32         // B_*H_
#define CHUNK 32         // K per wmma_f32_16x16x32_bf16
#define RPB   128        // query rows per block (8 waves * 16)
#define XPAD  40         // padded row stride (bf16) of transposed V tile (80B = 5*16B)
#define NEG_SLOPE 0.2f

typedef __attribute__((ext_vector_type(16))) __bf16 v16bf;
typedef __attribute__((ext_vector_type(8)))  __bf16 v8bf;
typedef __attribute__((ext_vector_type(8)))  float  v8f;
typedef __attribute__((ext_vector_type(2)))  float  v2f;

// ---- CDNA5 async copy: global -> LDS, 16B per lane, tracked by ASYNCcnt ----
__device__ __forceinline__ void async_ld16(uint32_t lds_off, const void* gptr) {
  asm volatile("global_load_async_to_lds_b128 %0, %1, off"
               :: "v"(lds_off), "v"((unsigned long long)(uintptr_t)gptr)
               : "memory");
}
__device__ __forceinline__ void wait_async0() {
  asm volatile("s_wait_asynccnt 0x0" ::: "memory");
}

// ---------------------------------------------------------------------------
// Kernel 1: projection + attention logits, fp32 WMMA (v_wmma_f32_16x16x4_f32).
// Block = 256 thr (8 waves) computes a 16-row x-tile against all 256 features
// (wave w -> feature tiles {w, w+8}).  Outputs:
//   xpbT[b*HF + f][n] = bf16(xp)        (feature-major, async-copyable V)
//   srcv/dstv[b,h,n]  = fp32 logits     (fixed-order reductions: deterministic)
// A operand per ISA 7.12.2 (32-bit A 16x4): lane il holds row il, elems
// K = {2*hi, 2*hi+1}; C: VGPR k <-> row k+8*hi, col il.
// ---------------------------------------------------------------------------
__global__ __launch_bounds__(256)
void gat_proj_kernel(const float* __restrict__ x,
                     const float* __restrict__ W,
                     const float* __restrict__ att_src,
                     const float* __restrict__ att_dst,
                     __bf16* __restrict__ xpbT,
                     float* __restrict__ srcv,
                     float* __restrict__ dstv) {
  __shared__ __align__(16) float sx[16][FIN];   // 8KB x tile (rows n0..n0+15)
  __shared__ float s_part[8][2][2][16];         // [wave][tile slot][src/dst][row]

  const int blk = blockIdx.x;
  const int b   = blk >> 7;                     // 128 row-tiles per batch
  const int n0  = (blk & 127) << 4;
  const int tid = threadIdx.x;
  const int wave = tid >> 5, lane = tid & 31;
  const int il = lane & 15, hi = lane >> 4;

#pragma unroll
  for (int i = 0; i < 8; ++i) {                 // coalesced x-tile load
    const int idx = tid + i * 256;              // 0..2047
    sx[idx >> 7][idx & 127] =
        x[((size_t)b * N_ + n0 + (idx >> 7)) * FIN + (idx & 127)];
  }
  __syncthreads();

#pragma unroll
  for (int slot = 0; slot < 2; ++slot) {
    const int tile = wave + slot * 8;           // feature tile 0..15
    const int fg   = tile * 16 + il;            // global feature 0..255
    const float* wr = W + (size_t)fg * FIN;
    v8f c = {};
#pragma unroll 8
    for (int k0 = 0; k0 < FIN; k0 += 4) {
      v2f a  = *(const v2f*)&sx[il][k0 + hi * 2];
      v2f bb = *(const v2f*)&wr[k0 + hi * 2];
      c = __builtin_amdgcn_wmma_f32_16x16x4_f32(false, a, false, bb,
                                                (short)0, c, false, false);
    }
    // feature-major bf16 store: lane holds 8 consecutive n for feature fg
    v8bf pk;
#pragma unroll
    for (int k = 0; k < 8; ++k) pk[k] = (__bf16)c[k];
    *(v8bf*)(xpbT + ((size_t)b * HF + fg) * N_ + n0 + hi * 8) = pk;

    // logit partials: reduce over the 16 features of this tile (fixed order)
    const float as = att_src[fg], ad = att_dst[fg];
#pragma unroll
    for (int k = 0; k < 8; ++k) {
      float vs = c[k] * as, vd = c[k] * ad;
#pragma unroll
      for (int off = 1; off < 16; off <<= 1) {
        vs += __shfl_xor(vs, off, 32);
        vd += __shfl_xor(vd, off, 32);
      }
      if (il == 0) {
        s_part[wave][slot][0][k + hi * 8] = vs;
        s_part[wave][slot][1][k + hi * 8] = vd;
      }
    }
  }
  __syncthreads();
  if (tid < 128) {                              // {src,dst} x 4 heads x 16 rows
    const int r  = tid & 15;
    const int h  = (tid >> 4) & 3;
    const int sd = tid >> 6;
    float sum = 0.f;
#pragma unroll
    for (int q = 0; q < 4; ++q) {               // 4 feature tiles per head
      const int t4 = h * 4 + q;
      sum += s_part[t4 & 7][t4 >> 3][sd][r];
    }
    (sd ? dstv : srcv)[((size_t)b * H_ + h) * N_ + n0 + r] = sum;
  }
}

// ---------------------------------------------------------------------------
// Kernel 2: fused masked-softmax attention, flash style, bf16 WMMA for P@V,
// double-buffered async-to-LDS staging (adj tile 16KB, V tile, dst logits).
// grid = (N/128, B*H); block = 256 (8 waves, 16 query rows per wave).
// ---------------------------------------------------------------------------
__global__ __launch_bounds__(256)
void gat_attn_kernel(const __bf16* __restrict__ xpbT,
                     const float* __restrict__ srcv,
                     const float* __restrict__ dstv,
                     const int* __restrict__ adj,
                     float* __restrict__ out) {
  __shared__ __align__(16) int    s_adj[2][RPB * CHUNK];   // 2 x 16KB
  __shared__ __align__(16) __bf16 s_xpT[2][FOUT * XPAD];   // 2 x 5KB, [f][j] pad 40
  __shared__ __align__(16) float  s_dst[2][CHUNK];

  const int bh   = blockIdx.y;
  const int b    = bh >> 2, h = bh & 3;
  const int row0 = blockIdx.x * RPB;
  const int tid  = threadIdx.x;
  const int wave = tid >> 5;
  const int lane = tid & 31;
  const int il   = lane & 15;
  const int hi   = lane >> 4;
  const int jbase = hi * 8;
  const int i0   = row0 + wave * 16;
  const int irow = i0 + il;

  const __bf16* xpTbh = xpbT + (size_t)bh * FOUT * N_;     // rows = features
  const float*  dbh   = dstv + (size_t)bh * N_;
  const int*    adjb  = adj  + (size_t)b * N_ * N_;
  const float   src_i = srcv[(size_t)bh * N_ + irow];

  const uint32_t adj_base = (uint32_t)(uintptr_t)&s_adj[0][0];
  const uint32_t xpT_base = (uint32_t)(uintptr_t)&s_xpT[0][0];
  const uint32_t dst_base = (uint32_t)(uintptr_t)&s_dst[0][0];

  // issue one chunk's staging as async 16B ops (all addresses 16B aligned)
  auto stage = [&](int jcn, int bufn) {
    const char* at = (const char*)(adjb + (size_t)row0 * N_ + jcn);
    const uint32_t ad = adj_base + bufn * (RPB * CHUNK * 4);
#pragma unroll
    for (int i = 0; i < 4; ++i) {               // adj: 128 rows x 128B
      const int u = tid + i * 256;              // 0..1023 16B units
      async_ld16(ad + u * 16, at + (size_t)(u >> 3) * (N_ * 4) + (u & 7) * 16);
    }
    {                                           // V tile: 64 rows x 64B
      const int f = tid >> 2, c = tid & 3;
      async_ld16(xpT_base + bufn * (FOUT * XPAD * 2) + f * (XPAD * 2) + c * 16,
                 (const char*)(xpTbh + (size_t)f * N_ + jcn) + c * 16);
    }
    if (tid < 8)                                // dst logits: 128B
      async_ld16(dst_base + bufn * (CHUNK * 4) + tid * 16,
                 (const char*)(dbh + jcn) + tid * 16);
  };

  v8f acc0 = {}, acc1 = {}, acc2 = {}, acc3 = {};
  float m = -INFINITY, l = 0.f;

  stage(0, 0);
  wait_async0();
  __syncthreads();

  for (int jc = 0; jc < N_; jc += CHUNK) {
    const int buf = (jc >> 5) & 1;
    if (jc + CHUNK < N_) stage(jc + CHUNK, buf ^ 1);  // prefetch overlaps compute

    // ---- scores for this lane's (row, 16 K-slots matching A layout) ----
    float p[16];
    float cmax = -INFINITY;
#pragma unroll
    for (int t = 0; t < 16; ++t) {
      const int k = (t < 8) ? (jbase + t) : (16 + jbase + (t - 8));
      const int a = s_adj[buf][(wave * 16 + il) * CHUNK + k];
      float e = src_i + s_dst[buf][k];
      e = (e >= 0.f) ? e : NEG_SLOPE * e;       // LeakyReLU(0.2)
      p[t] = a ? e : -INFINITY;
      cmax = fmaxf(cmax, p[t]);
    }
    cmax = fmaxf(cmax, __shfl_xor(cmax, 16, 32));
    const float mnew  = fmaxf(m, cmax);
    const float alpha = (mnew == -INFINITY) ? 1.f : __expf(m - mnew);

    float psum = 0.f;
    v16bf afrag;
#pragma unroll
    for (int t = 0; t < 16; ++t) {
      const float pe = (p[t] == -INFINITY) ? 0.f : __expf(p[t] - mnew);
      psum += pe;
      afrag[t] = (__bf16)pe;
    }
    psum += __shfl_xor(psum, 16, 32);
    l = l * alpha + psum;
    m = mnew;

    // rescale accumulators: alpha for row r lives in lane r (rows<->lanes 0..15)
#pragma unroll
    for (int k = 0; k < 8; ++k) {
      const float ak = __shfl(alpha, k + hi * 8, 32);
      acc0[k] *= ak; acc1[k] *= ak; acc2[k] *= ak; acc3[k] *= ak;
    }

    // B fragments: two contiguous 16B LDS loads per N-tile from padded [f][j]
    const int base0 = jbase;
    const int base1 = 16 + jbase;
#pragma unroll
    for (int nt = 0; nt < 4; ++nt) {
      const int n = nt * 16 + il;
      v8bf lo  = *(const v8bf*)(&s_xpT[buf][0] + n * XPAD + base0);
      v8bf hi8 = *(const v8bf*)(&s_xpT[buf][0] + n * XPAD + base1);
      v16bf bfrag = __builtin_shufflevector(lo, hi8, 0, 1, 2, 3, 4, 5, 6, 7,
                                            8, 9, 10, 11, 12, 13, 14, 15);
      switch (nt) {
        case 0: acc0 = __builtin_amdgcn_wmma_f32_16x16x32_bf16(false, afrag, false, bfrag, (short)0, acc0, false, false); break;
        case 1: acc1 = __builtin_amdgcn_wmma_f32_16x16x32_bf16(false, afrag, false, bfrag, (short)0, acc1, false, false); break;
        case 2: acc2 = __builtin_amdgcn_wmma_f32_16x16x32_bf16(false, afrag, false, bfrag, (short)0, acc2, false, false); break;
        case 3: acc3 = __builtin_amdgcn_wmma_f32_16x16x32_bf16(false, afrag, false, bfrag, (short)0, acc3, false, false); break;
      }
    }

    wait_async0();       // prefetch of next chunk has the whole compute to land
    __syncthreads();     // one barrier per chunk
  }

  // ---- epilogue: divide by row-sum l (per C-row broadcast), store [B,N,HF] ----
#pragma unroll
  for (int k = 0; k < 8; ++k) {
    const float lk = __shfl(l, k + hi * 8, 32);
    const float rl = (lk > 0.f) ? (1.f / lk) : 0.f;
    const int row  = i0 + k + hi * 8;
    float* orow = out + ((size_t)b * N_ + row) * HF + h * FOUT + il;
    orow[0]  = acc0[k] * rl;
    orow[16] = acc1[k] * rl;
    orow[32] = acc2[k] * rl;
    orow[48] = acc3[k] * rl;
  }
}

// ---------------------------------------------------------------------------
extern "C" void kernel_launch(void* const* d_in, const int* in_sizes, int n_in,
                              void* d_out, int out_size, void* d_ws, size_t ws_size,
                              hipStream_t stream) {
  (void)in_sizes; (void)n_in; (void)out_size; (void)ws_size;
  const float* x       = (const float*)d_in[0];
  const int*   adj     = (const int*)d_in[1];
  const float* W       = (const float*)d_in[2];
  const float* att_src = (const float*)d_in[3];
  const float* att_dst = (const float*)d_in[4];
  float* out = (float*)d_out;

  // workspace: xpbT (bf16, [B*HF, N]) | srcv | dstv  (~8.9 MB total)
  const size_t xpT_elems = (size_t)B_ * HF * N_;
  __bf16* xpbT = (__bf16*)d_ws;
  float* srcv = (float*)((char*)d_ws + xpT_elems * sizeof(__bf16));
  float* dstv = srcv + (size_t)BH * N_;

  gat_proj_kernel<<<dim3(B_ * (N_ / 16)), dim3(256), 0, stream>>>(
      x, W, att_src, att_dst, xpbT, srcv, dstv);
  gat_attn_kernel<<<dim3(N_ / RPB, BH), dim3(256), 0, stream>>>(
      xpbT, srcv, dstv, adj, out);
}
